// Network_85091892069069
// MI455X (gfx1250) — matrix operand order repaired and checked by reading, not compiled, not run on hardware
//
#include <hip/hip_runtime.h>
#include <hip/hip_bf16.h>

typedef _Float16 half_t;
typedef __attribute__((ext_vector_type(8)))  _Float16 v8h;
typedef __attribute__((ext_vector_type(16))) _Float16 v16h;
typedef __attribute__((ext_vector_type(8)))  float    v8f;

union V16cat { v16h v; v8h h[2]; };

static __device__ __forceinline__ v16h cat16(v8h lo, v8h hi) {
    V16cat u; u.h[0] = lo; u.h[1] = hi; return u.v;
}

// B fragment for V_WMMA_*_16x16x32_F16 (per K=32 slice):
// lane l holds column n = l&15; 16 contiguous K values, upper half-wave at +16.
// W is stored transposed in LDS: W[n*stride + k], k fastest, 16B-aligned rows.
static __device__ __forceinline__ v16h load_bfrag(const half_t* W, int stride, int n, int k0) {
    const v8h* p = (const v8h*)(W + n * stride + k0);
    return cat16(p[0], p[1]);
}

static __device__ __forceinline__ v8f wmma16(v16h a, v16h b, v8f c) {
    return __builtin_amdgcn_wmma_f32_16x16x32_f16(
        /*neg_a=*/false, a, /*neg_b=*/false, b,
        /*c_mod=*/(short)0, c, /*reuse_a=*/false, /*reuse_b=*/false);
}

#define W1_STRIDE 40   // 32 K + pad (80B rows, 16B aligned)
#define W_STRIDE  72   // 64 K + pad (144B rows, 16B aligned)
#define H_STRIDE  72
#define TILES     4    // 16-row tiles per wave
#define ROWS_PER_BLOCK (8 * 16 * TILES)   // 512

__global__ __launch_bounds__(256, 1)
void fused_mlp_kernel(const float* __restrict__ x,
                      const float* __restrict__ W1, const float* __restrict__ b1,
                      const float* __restrict__ W2, const float* __restrict__ b2,
                      const float* __restrict__ W3, const float* __restrict__ b3,
                      float* __restrict__ out) {
    __shared__ __align__(16) half_t sW1[64 * W1_STRIDE];
    __shared__ __align__(16) half_t sW2[64 * W_STRIDE];
    __shared__ __align__(16) half_t sW3[64 * W_STRIDE];
    __shared__ float sB1[64], sB2[64], sB3[64];
    __shared__ __align__(16) half_t sH[8][16 * H_STRIDE];   // per-wave 16x64 activation tile

    const int tid  = threadIdx.x;
    const int wave = tid >> 5;
    const int lane = tid & 31;

    // ---- Stage weights (transposed, f16) and biases into LDS (once per block) ----
    for (int i = tid; i < 32 * 64; i += 256) {
        int k = i >> 6, n = i & 63;
        sW1[n * W1_STRIDE + k] = (half_t)W1[i];
    }
    for (int i = tid; i < 64 * 64; i += 256) {
        int k = i >> 6, n = i & 63;
        sW2[n * W_STRIDE + k] = (half_t)W2[i];
        sW3[n * W_STRIDE + k] = (half_t)W3[i];
    }
    if (tid < 64) { sB1[tid] = b1[tid]; sB2[tid] = b2[tid]; sB3[tid] = b3[tid]; }
    __syncthreads();

    const int m   = lane & 15;          // A row / B-C-D column index for this lane
    const int hi4 = lane >> 4;          // half-wave select
    const int kb  = hi4 * 16;           // A-fragment K base

    // ---- Hoist ALL B fragments + biases into registers (reused across tiles) ----
    v16h B1f[4], B2f[4][2], B3f[4][2];
    float bias1[4], bias2[4], bias3[4];
    #pragma unroll
    for (int nt = 0; nt < 4; ++nt) {
        const int n = nt * 16 + m;
        B1f[nt]    = load_bfrag(sW1, W1_STRIDE, n, hi4 * 16);
        B2f[nt][0] = load_bfrag(sW2, W_STRIDE,  n, 0  + hi4 * 16);
        B2f[nt][1] = load_bfrag(sW2, W_STRIDE,  n, 32 + hi4 * 16);
        B3f[nt][0] = load_bfrag(sW3, W_STRIDE,  n, 0  + hi4 * 16);
        B3f[nt][1] = load_bfrag(sW3, W_STRIDE,  n, 32 + hi4 * 16);
        bias1[nt] = sB1[n];
        bias2[nt] = sB2[n];
        bias3[nt] = sB3[n];
    }

    half_t* h = sH[wave];
    const long base = (long)blockIdx.x * ROWS_PER_BLOCK + wave * (16 * TILES);

    for (int t = 0; t < TILES; ++t) {
        const long row0 = base + t * 16;
        const float* xr = x + (row0 + m) * 32;

        // Prefetch next tile's rows while this tile computes.
        if (t + 1 < TILES)
            __builtin_prefetch(x + (row0 + 16 + m) * 32, 0, 1);

        // ---- A1 fragment (16x32 f16) straight from global x (f32) ----
        v16h A1;
        #pragma unroll
        for (int j = 0; j < 8; ++j) {
            A1[j]     = (half_t)xr[kb + j];
            A1[8 + j] = (half_t)xr[kb + 8 + j];
        }

        // ---- Layer 1: [16x32]x[32x64] + b1, ReLU -> LDS h-tile ----
        #pragma unroll
        for (int nt = 0; nt < 4; ++nt) {
            v8f c = {};
            c = wmma16(A1, B1f[nt], c);
            #pragma unroll
            for (int r = 0; r < 8; ++r) {
                float v = c[r] + bias1[nt];
                v = v > 0.f ? v : 0.f;
                h[(r + hi4 * 8) * H_STRIDE + nt * 16 + m] = (half_t)v;
            }
        }

        // ---- Layer 2: gather A2 (16x64), 2 WMMAs per N-tile, ReLU -> h-tile ----
        {
            const v8h* rp = (const v8h*)(h + m * H_STRIDE);
            v16h A2a = cat16(rp[2 * hi4 + 0], rp[2 * hi4 + 1]);   // K 0..31
            v16h A2b = cat16(rp[2 * hi4 + 4], rp[2 * hi4 + 5]);   // K 32..63
            #pragma unroll
            for (int nt = 0; nt < 4; ++nt) {
                v8f c = {};
                c = wmma16(A2a, B2f[nt][0], c);
                c = wmma16(A2b, B2f[nt][1], c);
                #pragma unroll
                for (int r = 0; r < 8; ++r) {
                    float v = c[r] + bias2[nt];
                    v = v > 0.f ? v : 0.f;
                    h[(r + hi4 * 8) * H_STRIDE + nt * 16 + m] = (half_t)v;
                }
            }
        }

        // ---- Layer 3: gather A3, + b3 (no activation) -> global out (f32) ----
        {
            const v8h* rp = (const v8h*)(h + m * H_STRIDE);
            v16h A3a = cat16(rp[2 * hi4 + 0], rp[2 * hi4 + 1]);
            v16h A3b = cat16(rp[2 * hi4 + 4], rp[2 * hi4 + 5]);
            #pragma unroll
            for (int nt = 0; nt < 4; ++nt) {
                v8f c = {};
                c = wmma16(A3a, B3f[nt][0], c);
                c = wmma16(A3b, B3f[nt][1], c);
                #pragma unroll
                for (int r = 0; r < 8; ++r) {
                    long rr = row0 + r + hi4 * 8;
                    out[rr * 64 + nt * 16 + m] = c[r] + bias3[nt];
                }
            }
        }
    }
}

extern "C" void kernel_launch(void* const* d_in, const int* in_sizes, int n_in,
                              void* d_out, int out_size, void* d_ws, size_t ws_size,
                              hipStream_t stream) {
    (void)n_in; (void)d_ws; (void)ws_size; (void)out_size;
    const float* x  = (const float*)d_in[0];
    const float* W1 = (const float*)d_in[1];
    const float* b1 = (const float*)d_in[2];
    const float* W2 = (const float*)d_in[3];
    const float* b2 = (const float*)d_in[4];
    const float* W3 = (const float*)d_in[5];
    const float* b3 = (const float*)d_in[6];
    float* out = (float*)d_out;

    const int n = in_sizes[0] / 32;   // sample rows
    dim3 grid((n + ROWS_PER_BLOCK - 1) / ROWS_PER_BLOCK);
    fused_mlp_kernel<<<grid, 256, 0, stream>>>(x, W1, b1, W2, b2, W3, b3, out);
}
